// elmo_ncrf_recipes_70016556859465
// MI455X (gfx1250) — matrix operand order repaired and checked by reading, not compiled, not run on hardware
//
#include <hip/hip_runtime.h>
#include <hip/hip_bf16.h>
#include <math.h>

typedef __attribute__((ext_vector_type(16))) __bf16 bfx16;
typedef __attribute__((ext_vector_type(4)))  __bf16 bfx4;
typedef __attribute__((ext_vector_type(8)))  float  fx8;

#define T_TOK   8192
#define D_IN    1024
#define H_DIM   1024
#define HH      512
#define G4      2048
#define NSTEP   256
#define SLEN    32
#define NING    64
#define NTAG    10
#define NEGV    -10000.0f

#define LROWS   256     // gate rows per scan block (4 gates x 64 units)
#define LSTR    520     // padded LDS row stride in bf16 (1040B, 16B aligned)

// ---------------- f32 -> bf16 conversion (4 elems/thread) ----------------
__global__ void cvt_bf16_kernel(const float* __restrict__ src, __bf16* __restrict__ dst, int n4){
  int i = blockIdx.x * blockDim.x + threadIdx.x;
  if (i < n4){
    float4 v = ((const float4*)src)[i];
    bfx4 o;
    o[0] = (__bf16)v.x; o[1] = (__bf16)v.y; o[2] = (__bf16)v.z; o[3] = (__bf16)v.w;
    ((bfx4*)dst)[i] = o;
  }
}

// ---------------- GEMM: C[M,N] = A[M,K] * W[N,K]^T + bias[N] (bf16 WMMA, f32 acc) ----
// grid = (N/64, ceil(M/128)), block = 256 (8 waves); wave w owns a 16x64 strip.
__global__ void gemm_wmma_kernel(const __bf16* __restrict__ A, const __bf16* __restrict__ W,
                                 const float* __restrict__ bias, float* __restrict__ C,
                                 int M, int N, int K){
  const int lane = threadIdx.x & 31;
  const int wave = threadIdx.x >> 5;
  const int l16  = lane & 15;
  const int half = lane >> 4;
  const int n0 = blockIdx.x * 64;
  const int m0 = blockIdx.y * 128 + wave * 16;
  if (m0 >= M) return;            // wave-uniform guard (e.g. M == 64 ingredient GEMM)
  const int m = m0 + l16;         // A-fragment row for this lane
  fx8 acc0 = {}, acc1 = {}, acc2 = {}, acc3 = {};
  const __bf16* arow = A + (size_t)m * K + 16 * half;
  const __bf16* w0r = W + (size_t)(n0      + l16) * K + 16 * half;
  const __bf16* w1r = W + (size_t)(n0 + 16 + l16) * K + 16 * half;
  const __bf16* w2r = W + (size_t)(n0 + 32 + l16) * K + 16 * half;
  const __bf16* w3r = W + (size_t)(n0 + 48 + l16) * K + 16 * half;
  for (int k0 = 0; k0 < K; k0 += 32){
    // A 16x32 bf16 fragment (contiguous k order, matching B): lanes 0-15 hold K=[0..15] of row m,
    // lanes 16-31 hold K=[16..31].
    bfx16 a;
    const __bf16* ap = arow + k0;
    #pragma unroll
    for (int i = 0; i < 8; i++){ a[i] = ap[i]; a[8 + i] = ap[8 + i]; }
    // B 32x16 fragments: contiguous 32B per lane from W row (n), half selects K range.
    bfx16 b0 = *(const bfx16*)(w0r + k0);
    bfx16 b1 = *(const bfx16*)(w1r + k0);
    bfx16 b2 = *(const bfx16*)(w2r + k0);
    bfx16 b3 = *(const bfx16*)(w3r + k0);
    acc0 = __builtin_amdgcn_wmma_f32_16x16x32_bf16(false, a, false, b0, (short)0, acc0, false, false);
    acc1 = __builtin_amdgcn_wmma_f32_16x16x32_bf16(false, a, false, b1, (short)0, acc1, false, false);
    acc2 = __builtin_amdgcn_wmma_f32_16x16x32_bf16(false, a, false, b2, (short)0, acc2, false, false);
    acc3 = __builtin_amdgcn_wmma_f32_16x16x32_bf16(false, a, false, b3, (short)0, acc3, false, false);
  }
  #pragma unroll
  for (int nn = 0; nn < 4; nn++){
    const fx8 cc = (nn == 0) ? acc0 : (nn == 1) ? acc1 : (nn == 2) ? acc2 : acc3;
    const int n = n0 + nn * 16 + l16;
    const float bb = bias[n];
    #pragma unroll
    for (int r = 0; r < 8; r++)
      C[(size_t)(m0 + r + 8 * half) * N + n] = cc[r] + bb;
  }
}

// ---------------- LSTM scan, sharded over 8 blocks via global arrive counter ----------
__device__ __forceinline__ float sigm(float x){ return 1.0f / (1.0f + expf(-x)); }

__device__ __forceinline__ void grid_bar(unsigned* bar, unsigned target){
  __syncthreads();
  if (threadIdx.x == 0){
    __threadfence();
    atomicAdd(bar, 1u);
    while (atomicAdd(bar, 0u) < target) __builtin_amdgcn_s_sleep(2);
    __threadfence();
  }
  __syncthreads();
}

#if defined(__gfx1250__) && __has_builtin(__builtin_amdgcn_tensor_load_to_lds)
#define USE_TDM 1
typedef __attribute__((ext_vector_type(4))) unsigned int u32x4;
typedef __attribute__((ext_vector_type(8))) int i32x8;
typedef __attribute__((ext_vector_type(4))) int i32x4;
#endif

// pre: [T][2048] gate preactivations (bias folded in by GEMM).
// Whh: [2048][512] bf16.  Block b owns hidden units [64b, 64b+64) and their 4 gate rows.
// Block = 512 threads: thread (r = tid&255, kh = tid>>8) computes the kh-half partial dot
// of gate row r.  hbuf: [2][512] ping-pong h broadcast.  out written at colOff..+511.
__global__ void lstm_scan_kernel(const float* __restrict__ pre, const __bf16* __restrict__ Whh,
                                 const float* __restrict__ h0, const float* __restrict__ c0,
                                 float* __restrict__ hbuf, float* __restrict__ out,
                                 int T, int back, int colOff, unsigned* __restrict__ bar){
  extern __shared__ char smem[];
  __bf16* Wsh = (__bf16*)smem;                               // LROWS * LSTR bf16 (260 KB)
  float*  hsh = (float*)(smem + (size_t)LROWS * LSTR * 2);   // 512
  float*  gsh = hsh + HH;                                    // 512 partials
  float*  csh = gsh + 2 * LROWS;                             // 64
  const int tid  = threadIdx.x;   // 512
  const int b    = blockIdx.x;    // 8
  const int r    = tid & 255;     // local gate row
  const int kh   = tid >> 8;      // k half (0/1)
  const int gidx = r >> 6;        // 0=i 1=f 2=g 3=o
  const int ul   = r & 63;        // local unit
  const int unit = b * 64 + ul;
  const int grow = gidx * HH + unit;  // global gate row

  // ---- Stage W_hh gate-row chunks into LDS ----
#ifdef USE_TDM
  if (tid < 32){   // wave 0 drives the Tensor Data Mover: 4 tiles of 64 rows x 512 bf16,
                   // padded 4 DWORDs every 256 DWORDs => LDS row stride 520 bf16.
    const unsigned ldsBase = (unsigned)(size_t)(void*)Wsh;
    #pragma unroll
    for (int g = 0; g < 4; g++){
      unsigned long long ga =
          (unsigned long long)(size_t)(Whh + ((size_t)g * HH + (size_t)b * 64) * HH);
      u32x4 g0;
      g0[0] = 1u;                                            // count=1 user descriptor
      g0[1] = ldsBase + (unsigned)(g * 64 * LSTR * 2);       // lds_addr
      g0[2] = (unsigned)(ga & 0xffffffffu);                  // global_addr[31:0]
      g0[3] = (unsigned)((ga >> 32) & 0x01ffffffu) | 0x80000000u;  // addr[56:32] | type=2
      i32x8 g1;
      g1[0] = (int)((1u << 16) |        // data_size = 2 bytes
                    (1u << 20) |        // pad_enable
                    (7u << 22) |        // pad_interval: 256 DWORDs
                    (3u << 25));        // pad_amount: 4 DWORDs
      g1[1] = (int)(512u << 16);        // tensor_dim0 = 512  (lo16 at bits 63:48)
      g1[2] = (int)(64u  << 16);        // tensor_dim0 hi = 0; tensor_dim1 = 64
      g1[3] = (int)(512u << 16);        // tensor_dim1 hi = 0; tile_dim0 = 512
      g1[4] = 64;                       // tile_dim1 = 64; tile_dim2 = 0
      g1[5] = 512;                      // tensor_dim0_stride lo32 = 512
      g1[6] = 0;
      g1[7] = 0;
      i32x4 z4 = {0, 0, 0, 0};
#if __clang_major__ >= 23
      i32x8 z8 = {0, 0, 0, 0, 0, 0, 0, 0};
      __builtin_amdgcn_tensor_load_to_lds(g0, g1, z4, z4, z8, 0);
#else
      __builtin_amdgcn_tensor_load_to_lds(g0, g1, z4, z4, 0);
#endif
    }
    __builtin_amdgcn_s_wait_tensorcnt(0);
  }
#else
  {
    const int row = tid >> 1, rh = tid & 1;       // 512 threads, 256 rows, 2 halves
    const int growR = (row >> 6) * HH + b * 64 + (row & 63);
    const uint4* src = (const uint4*)(Whh + (size_t)growR * HH + rh * 256);
    uint4* dst = (uint4*)(Wsh + (size_t)row * LSTR + rh * 256);
    #pragma unroll 8
    for (int i = 0; i < 32; i++) dst[i] = src[i];
  }
#endif
  if (tid < 64){
    csh[tid] = c0[b * 64 + tid];
    hbuf[b * 64 + tid] = h0[b * 64 + tid];
  }
  unsigned ph = 1;
  grid_bar(bar, 8u * ph); ph++;
  for (int ti = 0; ti < T; ti++){
    const int t   = back ? (T - 1 - ti) : ti;
    const int cur = ti & 1;
    if (tid < HH) hsh[tid] = hbuf[cur * HH + tid];
    __syncthreads();
    // half-row partial: dot(h[256*kh .. +256), Whh_row[r][256*kh .. +256))
    float acc = kh ? 0.f : pre[(size_t)t * G4 + grow];
    const unsigned* wrow = (const unsigned*)(Wsh + (size_t)r * LSTR + 256 * kh);
    const float* hs = hsh + 256 * kh;
    #pragma unroll 8
    for (int kk = 0; kk < 128; kk++){
      unsigned wp = wrow[kk];
      float w0 = __uint_as_float(wp << 16);
      float w1 = __uint_as_float(wp & 0xffff0000u);
      acc = fmaf(hs[2 * kk],     w0, acc);
      acc = fmaf(hs[2 * kk + 1], w1, acc);
    }
    gsh[kh * LROWS + r] = acc;
    __syncthreads();
    if (tid < 64){
      float ig = gsh[tid]       + gsh[LROWS + tid];
      float fg = gsh[64  + tid] + gsh[LROWS + 64  + tid];
      float gg = gsh[128 + tid] + gsh[LROWS + 128 + tid];
      float og = gsh[192 + tid] + gsh[LROWS + 192 + tid];
      float cn = sigm(fg) * csh[tid] + sigm(ig) * tanhf(gg);
      float hn = sigm(og) * tanhf(cn);
      csh[tid] = cn;
      hbuf[(cur ^ 1) * HH + unit] = hn;
      out[(size_t)t * H_DIM + colOff + unit] = hn;
    }
    grid_bar(bar, 8u * ph); ph++;
  }
}

// ---------------- per-step attention (only ingredient column 0 survives) -------------
__global__ void attn_step_kernel(const float* __restrict__ lstm_out,
                                 const float* __restrict__ aing0,   // attn_ing row 0 [1024]
                                 float* __restrict__ step_feats){
  __shared__ float part[256];
  __shared__ float p_sh[SLEN];
  const int s = blockIdx.x, tid = threadIdx.x;
  const int l = tid >> 3, seg = tid & 7;   // 8 segments x 128 per token dot
  {
    const float* row = lstm_out + ((size_t)s * SLEN + l) * H_DIM + seg * 128;
    const float* a = aing0 + seg * 128;
    float acc = 0.f;
    #pragma unroll 4
    for (int k = 0; k < 128; k++) acc = fmaf(row[k], a[k], acc);
    part[tid] = acc;
  }
  __syncthreads();
  if (tid < SLEN){
    float sc = 0.f;
    #pragma unroll
    for (int i = 0; i < 8; i++) sc += part[tid * 8 + i];
    p_sh[tid] = sc;
  }
  __syncthreads();
  if (tid == 0){
    float m = p_sh[0];
    for (int i = 1; i < SLEN; i++) m = fmaxf(m, p_sh[i]);
    float sum = 0.f;
    for (int i = 0; i < SLEN; i++){ float e = expf(p_sh[i] - m); p_sh[i] = e; sum += e; }
    float inv = 1.0f / sum;
    for (int i = 0; i < SLEN; i++) p_sh[i] *= inv;
  }
  __syncthreads();
  for (int j = tid; j < H_DIM; j += 256){
    const float* base = lstm_out + (size_t)s * SLEN * H_DIM + j;
    float acc = 0.f;
    #pragma unroll 8
    for (int l2 = 0; l2 < SLEN; l2++) acc = fmaf(p_sh[l2], base[(size_t)l2 * H_DIM], acc);
    step_feats[(size_t)s * H_DIM + j] = acc;
  }
}

// ---------------- emission features: feats[s][tag] = ent[s].W_tag[tag] + b_tag --------
__global__ void feats_kernel(const float* __restrict__ ent, const float* __restrict__ Wtag,
                             const float* __restrict__ btag, float* __restrict__ feats){
  const int s = blockIdx.x;
  const int tag = threadIdx.x >> 5, lane = threadIdx.x & 31;
  if (tag >= NTAG) return;   // wave-uniform exit
  const float* e = ent + (size_t)s * H_DIM;
  const float* w = Wtag + (size_t)tag * H_DIM;
  float acc = 0.f;
  for (int k = lane; k < H_DIM; k += 32) acc = fmaf(e[k], w[k], acc);
  for (int o = 16; o > 0; o >>= 1) acc += __shfl_down(acc, o, 32);
  if (lane == 0) feats[s * NTAG + tag] = acc + btag[tag];
}

// ---------------- CRF forward + gold path, NLL to out[0] ------------------------------
__global__ void crf_kernel(const float* __restrict__ feats, const float* __restrict__ trans,
                           const int* __restrict__ targets, float* __restrict__ out){
  __shared__ float tr[NTAG * NTAG];
  __shared__ float alpha[NTAG];
  __shared__ float nalpha[NTAG];
  const int tid = threadIdx.x;
  if (tid < NTAG * NTAG) tr[tid] = trans[tid];
  if (tid < NTAG) alpha[tid] = (tid == NTAG - 2) ? 0.f : NEGV;
  __syncthreads();
  for (int s = 0; s < NSTEP; s++){
    if (tid < NTAG){
      float m = -3.4e38f;
      #pragma unroll
      for (int i = 0; i < NTAG; i++) m = fmaxf(m, alpha[i] + tr[tid * NTAG + i]);
      float sum = 0.f;
      #pragma unroll
      for (int i = 0; i < NTAG; i++) sum += expf(alpha[i] + tr[tid * NTAG + i] - m);
      nalpha[tid] = logf(sum) + m + feats[s * NTAG + tid];
    }
    __syncthreads();
    if (tid < NTAG) alpha[tid] = nalpha[tid];
    __syncthreads();
  }
  if (tid == 0){
    float m = -3.4e38f;
    for (int i = 0; i < NTAG; i++) m = fmaxf(m, alpha[i] + tr[(NTAG - 1) * NTAG + i]);
    float sum = 0.f;
    for (int i = 0; i < NTAG; i++) sum += expf(alpha[i] + tr[(NTAG - 1) * NTAG + i] - m);
    float logz = logf(sum) + m;
    float gold = 0.f;
    int prev = NTAG - 2;
    for (int s = 0; s < NSTEP; s++){
      int tg = targets[s];
      gold += feats[s * NTAG + tg] + tr[tg * NTAG + prev];
      prev = tg;
    }
    gold += tr[(NTAG - 1) * NTAG + targets[NSTEP - 1]];
    out[0] = logz - gold;
  }
}

// =====================================================================================
extern "C" void kernel_launch(void* const* d_in, const int* in_sizes, int n_in,
                              void* d_out, int out_size, void* d_ws, size_t ws_size,
                              hipStream_t stream){
  (void)in_sizes; (void)n_in; (void)out_size; (void)ws_size;
  const float* elmo_sent = (const float*)d_in[0];
  const float* elmo_ings = (const float*)d_in[1];
  const float* Ws_ih_f = (const float*)d_in[2];
  const float* Ws_hh_f = (const float*)d_in[3];
  const float* bs_f    = (const float*)d_in[4];
  const float* Ws_ih_b = (const float*)d_in[5];
  const float* Ws_hh_b = (const float*)d_in[6];
  const float* bs_b    = (const float*)d_in[7];
  const float* We_ih_f = (const float*)d_in[8];
  const float* We_hh_f = (const float*)d_in[9];
  const float* be_f    = (const float*)d_in[10];
  const float* We_ih_b = (const float*)d_in[11];
  const float* We_hh_b = (const float*)d_in[12];
  const float* be_b    = (const float*)d_in[13];
  const float* W_attn  = (const float*)d_in[14];
  const float* b_attn  = (const float*)d_in[15];
  const float* W_tag   = (const float*)d_in[16];
  const float* b_tag   = (const float*)d_in[17];
  const float* trans   = (const float*)d_in[18];
  const float* h0_sent = (const float*)d_in[19];
  const float* c0_sent = (const float*)d_in[20];
  const float* h0_ent  = (const float*)d_in[21];
  const float* c0_ent  = (const float*)d_in[22];
  const int*   targets = (const int*)d_in[23];

  // ---- workspace bump allocator (256B aligned) ----
  char* ws = (char*)d_ws;
  size_t off = 0;
  auto alloc = [&](size_t bytes) -> void* {
    void* p = ws + off;
    off = (off + bytes + 255) & ~(size_t)255;
    return p;
  };
  __bf16* xs_bf    = (__bf16*)alloc((size_t)T_TOK * D_IN * 2);
  __bf16* wihf_bf  = (__bf16*)alloc((size_t)G4 * D_IN * 2);
  __bf16* wihb_bf  = (__bf16*)alloc((size_t)G4 * D_IN * 2);
  __bf16* whhf_bf  = (__bf16*)alloc((size_t)G4 * HH * 2);
  __bf16* whhb_bf  = (__bf16*)alloc((size_t)G4 * HH * 2);
  __bf16* weihf_bf = (__bf16*)alloc((size_t)G4 * H_DIM * 2);
  __bf16* weihb_bf = (__bf16*)alloc((size_t)G4 * H_DIM * 2);
  __bf16* wehhf_bf = (__bf16*)alloc((size_t)G4 * HH * 2);
  __bf16* wehhb_bf = (__bf16*)alloc((size_t)G4 * HH * 2);
  __bf16* wattn_bf = (__bf16*)alloc((size_t)H_DIM * D_IN * 2);
  __bf16* ings_bf  = (__bf16*)alloc((size_t)NING * D_IN * 2);
  __bf16* sf_bf    = (__bf16*)alloc((size_t)NSTEP * H_DIM * 2);
  float* pre_f     = (float*)alloc((size_t)T_TOK * G4 * 4);
  float* pre_b     = (float*)alloc((size_t)T_TOK * G4 * 4);
  float* lstm_out  = (float*)alloc((size_t)T_TOK * H_DIM * 4);
  float* attn_ing  = (float*)alloc((size_t)NING * H_DIM * 4);
  float* step_feats= (float*)alloc((size_t)NSTEP * H_DIM * 4);
  float* pre_ef    = (float*)alloc((size_t)NSTEP * G4 * 4);
  float* pre_eb    = (float*)alloc((size_t)NSTEP * G4 * 4);
  float* ent       = (float*)alloc((size_t)NSTEP * H_DIM * 4);
  float* feats     = (float*)alloc((size_t)NSTEP * NTAG * 4);
  float* hb0       = (float*)alloc(2 * HH * 4);
  float* hb1       = (float*)alloc(2 * HH * 4);
  float* hb2       = (float*)alloc(2 * HH * 4);
  float* hb3       = (float*)alloc(2 * HH * 4);
  unsigned* bars   = (unsigned*)alloc(256);

  hipMemsetAsync(bars, 0, 256, stream);

  auto cvt = [&](const float* s, __bf16* d, int n){
    int n4 = n / 4;
    cvt_bf16_kernel<<<(n4 + 255) / 256, 256, 0, stream>>>(s, d, n4);
  };
  cvt(elmo_sent, xs_bf,    T_TOK * D_IN);
  cvt(Ws_ih_f,  wihf_bf,   G4 * D_IN);
  cvt(Ws_ih_b,  wihb_bf,   G4 * D_IN);
  cvt(Ws_hh_f,  whhf_bf,   G4 * HH);
  cvt(Ws_hh_b,  whhb_bf,   G4 * HH);
  cvt(We_ih_f,  weihf_bf,  G4 * H_DIM);
  cvt(We_ih_b,  weihb_bf,  G4 * H_DIM);
  cvt(We_hh_f,  wehhf_bf,  G4 * HH);
  cvt(We_hh_b,  wehhb_bf,  G4 * HH);
  cvt(W_attn,   wattn_bf,  H_DIM * D_IN);
  cvt(elmo_ings, ings_bf,  NING * D_IN);

  auto gemm = [&](const __bf16* A, const __bf16* W, const float* bias, float* C,
                  int M, int N, int K){
    dim3 g(N / 64, (M + 127) / 128);
    gemm_wmma_kernel<<<g, 256, 0, stream>>>(A, W, bias, C, M, N, K);
  };
  // sentence LSTM input projections: pre = x @ W_ih^T + b  [8192, 2048]
  gemm(xs_bf, wihf_bf, bs_f, pre_f, T_TOK, G4, D_IN);
  gemm(xs_bf, wihb_bf, bs_b, pre_b, T_TOK, G4, D_IN);
  // ingredient projection: attn_ing = ings @ W_attn^T + b_attn  [64, 1024]
  gemm(ings_bf, wattn_bf, b_attn, attn_ing, NING, H_DIM, D_IN);

  // sentence BiLSTM scans (8 blocks sharing W_hh in LDS, global arrive barrier)
  const size_t SCAN_LDS = (size_t)LROWS * LSTR * 2 + (HH + 2 * LROWS + 64) * sizeof(float);
  hipFuncSetAttribute((const void*)lstm_scan_kernel,
                      hipFuncAttributeMaxDynamicSharedMemorySize, (int)SCAN_LDS);
  lstm_scan_kernel<<<8, 512, SCAN_LDS, stream>>>(pre_f, whhf_bf, h0_sent, c0_sent,
                                                 hb0, lstm_out, T_TOK, 0, 0,  bars + 0);
  lstm_scan_kernel<<<8, 512, SCAN_LDS, stream>>>(pre_b, whhb_bf, h0_sent + HH, c0_sent + HH,
                                                 hb1, lstm_out, T_TOK, 1, HH, bars + 16);

  // per-step attention pooling (ingredient 0 column only)
  attn_step_kernel<<<NSTEP, 256, 0, stream>>>(lstm_out, attn_ing, step_feats);

  // entity LSTM input projections
  cvt(step_feats, sf_bf, NSTEP * H_DIM);
  gemm(sf_bf, weihf_bf, be_f, pre_ef, NSTEP, G4, H_DIM);
  gemm(sf_bf, weihb_bf, be_b, pre_eb, NSTEP, G4, H_DIM);

  // entity BiLSTM scans
  lstm_scan_kernel<<<8, 512, SCAN_LDS, stream>>>(pre_ef, wehhf_bf, h0_ent, c0_ent,
                                                 hb2, ent, NSTEP, 0, 0,  bars + 32);
  lstm_scan_kernel<<<8, 512, SCAN_LDS, stream>>>(pre_eb, wehhb_bf, h0_ent + HH, c0_ent + HH,
                                                 hb3, ent, NSTEP, 1, HH, bars + 48);

  // emissions + CRF NLL
  feats_kernel<<<NSTEP, 512, 0, stream>>>(ent, W_tag, b_tag, feats);
  crf_kernel<<<1, 128, 0, stream>>>(feats, trans, targets, (float*)d_out);
}
